// node_aggregation_28647431864536
// MI455X (gfx1250) — matrix operand order repaired and checked by reading, not compiled, not run on hardware
//
#include <hip/hip_runtime.h>
#include <hip/hip_bf16.h>
#include <stdint.h>

// Problem constants (fixed by the reference)
#define NN     50000
#define EE     400000
#define INF_   256
#define HID    1000
#define OUTF   128

#define MPAD   50048      // 16 * 3128  (3128 divisible by 4 for GEMM2 WPM=4)
#define HIDPAD 1024       // GEMM1 N padded (multiple of 64*WPN)
#define KPAD2  1024       // GEMM2 K padded (multiple of 32)

typedef __attribute__((ext_vector_type(16))) __bf16 v16bf;
typedef __attribute__((ext_vector_type(8)))  float  v8f;

struct alignas(16) Q16 { unsigned int w0, w1, w2, w3; };
union Frag { v16bf v; Q16 q[2]; };

__device__ __forceinline__ __bf16 f2bf(float f) { return (__bf16)f; }

// ---------------------------------------------------------------------------
// CDNA5 async copy (global -> LDS, tracked by ASYNCcnt)
// ---------------------------------------------------------------------------
typedef int v4i_vs __attribute__((vector_size(16)));

__device__ __forceinline__ void async_copy_b128(void* lds_ptr, const void* gptr) {
#if __has_builtin(__builtin_amdgcn_global_load_async_to_lds_b128)
  __builtin_amdgcn_global_load_async_to_lds_b128(
      (__attribute__((address_space(1))) v4i_vs*)(uintptr_t)gptr,
      (__attribute__((address_space(3))) v4i_vs*)lds_ptr,
      0, 0);
#else
  unsigned int off = (unsigned int)(uintptr_t)(
      (__attribute__((address_space(3))) char*)lds_ptr);
  asm volatile("global_load_async_to_lds_b128 %0, %1, off"
               :: "v"(off), "v"(gptr) : "memory");
#endif
}

__device__ __forceinline__ void wait_async0() {
#if __has_builtin(__builtin_amdgcn_s_wait_asynccnt)
  __builtin_amdgcn_s_wait_asynccnt(0);
#else
  asm volatile("s_wait_asynccnt 0x0" ::: "memory");
#endif
}

__device__ __forceinline__ void wait_tensor0() {
#if __has_builtin(__builtin_amdgcn_s_wait_tensorcnt)
  __builtin_amdgcn_s_wait_tensorcnt(0);
#else
  asm volatile("s_wait_tensorcnt 0x0" ::: "memory");
#endif
}

// ---------------------------------------------------------------------------
// CDNA5 Tensor Data Mover: 2D tile global -> LDS (tracked by TENSORcnt)
// D# per CDNA5 ISA. 2-byte elements. LDS padding:
// 16B inserted after every 64B row -> LDS row pitch 80B (40 elements).
// This toolchain exposes the 6-arg builtin:
//   (uint32x4 g0, int32x8 g1, int32x4 g2, int32x4 g3, int32x8, int cpol)
// ---------------------------------------------------------------------------
#if __has_builtin(__builtin_amdgcn_tensor_load_to_lds)
#define HAVE_TDM 1
typedef unsigned int u32x4_vs __attribute__((vector_size(16)));
typedef int          i32x8_vs __attribute__((vector_size(32)));
typedef int          i32x4_vs __attribute__((vector_size(16)));

__device__ __forceinline__ void tdm_load_2d_bf16(
    unsigned lds_off_bytes, const void* gaddr,
    unsigned tensor_d0 /*K elems*/, unsigned tensor_d1 /*rows*/,
    unsigned tile_d0 /*K elems*/, unsigned tile_d1 /*rows*/,
    unsigned stride_d0 /*K elems*/) {
  unsigned long long ga = (unsigned long long)(uintptr_t)gaddr;
  u32x4_vs g0;
  g0[0] = 1u;                                    // count=1, user descriptor
  g0[1] = lds_off_bytes;                         // lds_addr (bytes)
  g0[2] = (unsigned)(ga & 0xFFFFFFFFu);          // global_addr[31:0]
  g0[3] = (unsigned)((ga >> 32) & 0x01FFFFFFu)   // global_addr[56:32]
        | (2u << 30);                            // type = 2 ("image")
  i32x8_vs g1;
  g1[0] = (int)((1u << 16)    // data_size = 1 -> 2 bytes
              | (1u << 20)    // pad_enable
              | (3u << 22)    // pad_interval: 16 DWORDs (one 64B row)
              | (3u << 25));  // pad_amount: 4 DWORDs (16B)
  g1[1] = (int)((tensor_d0 & 0xFFFFu) << 16);    // tensor_dim0[15:0]
  g1[2] = (int)(((tensor_d0 >> 16) & 0xFFFFu)    // tensor_dim0[31:16]
              | ((tensor_d1 & 0xFFFFu) << 16));  // tensor_dim1[15:0]
  g1[3] = (int)(((tensor_d1 >> 16) & 0xFFFFu)    // tensor_dim1[31:16]
              | ((tile_d0 & 0xFFFFu) << 16));    // tile_dim0
  g1[4] = (int)(tile_d1 & 0xFFFFu);              // tile_dim1 (tile_dim2=0)
  g1[5] = (int)stride_d0;                        // tensor_dim0_stride[31:0]
  g1[6] = 0;                                     // stride hi | dim1_stride lo
  g1[7] = 0;
  i32x4_vs z4 = {0, 0, 0, 0};
  i32x8_vs z8 = {0, 0, 0, 0, 0, 0, 0, 0};
  __builtin_amdgcn_tensor_load_to_lds(g0, g1, z4, z4, z8, 0);
}
#endif

// ---------------------------------------------------------------------------
// WMMA GEMM with double-buffered LDS A staging (async-copy or TDM) and
// register double-buffered B fragments.
//   C[M x Nstore] = A[M x K] * B[K x Npad], BT is [Npad x K] (K-major cols).
// Block = WPM*WPN waves (8). Block tile = (16*WPM) x (64*WPN).
// LDS A tile row pitch = 40 elements (80B) to de-conflict ds_load_b128 banks
// and match the TDM pad settings.
// ---------------------------------------------------------------------------
template <int WPM, int WPN, bool UseTDM, typename OutT>
__global__ __launch_bounds__(32 * WPM * WPN)
void gemm_bf16_async(const __bf16* __restrict__ A,
                     const __bf16* __restrict__ BT,
                     OutT* __restrict__ C,
                     int M, int Npad, int Nstore, int K, int ldc) {
  constexpr int LROW   = 40;            // LDS row pitch in elements
  constexpr int TILE_E = 16 * LROW;     // one 16-row tile
  constexpr int BUF_E  = WPM * TILE_E;  // one k-step buffer
  __shared__ __align__(16) __bf16 As[2 * BUF_E];

  const int tid  = threadIdx.x;
  const int lane = tid & 31;
  const int w    = tid >> 5;
  const int wm   = w / WPN;
  const int wn   = w % WPN;

  const int nbn = Npad / (64 * WPN);
  const int bm  = (int)blockIdx.x / nbn;
  const int bn  = (int)blockIdx.x % nbn;

  const int rowB = bm * 16 * WPM;            // block row base
  const int row0 = rowB + wm * 16;           // wave row base
  const int col0 = bn * 64 * WPN + wn * 64;  // wave col base

  const int half = lane >> 4;
  const int mr   = lane & 15;

  // Async staging assignment: chunk -> (tile st, row sr, 16B part sp)
  const bool stager = (tid < WPM * 64);
  const int st = tid >> 6;
  const int ci = tid & 63;
  const int sr = ci >> 2;
  const int sp = ci & 3;
  const __bf16* gsrc = A + (size_t)(rowB + st * 16 + sr) * K + sp * 8;

#if defined(HAVE_TDM)
  const unsigned asbase = (unsigned)(uintptr_t)(
      (__attribute__((address_space(3))) char*)&As[0]);
#endif

  // Stage k-step s into buffer buf.
  auto stage = [&](int k0, int buf) {
    if constexpr (UseTDM) {
#if defined(HAVE_TDM)
      if (w == 0)
        tdm_load_2d_bf16(asbase + (unsigned)buf * BUF_E * 2,
                         A + (size_t)rowB * K + k0,
                         (unsigned)K, (unsigned)M, 32u, (unsigned)(16 * WPM),
                         (unsigned)K);
      return;
#endif
    }
    if (stager)
      async_copy_b128(&As[buf * BUF_E + st * TILE_E + sr * LROW + sp * 8],
                      gsrc + k0);
  };
  auto stage_wait = [&]() {
    if constexpr (UseTDM) {
#if defined(HAVE_TDM)
      wait_tensor0();
      return;
#endif
    }
    wait_async0();
  };

  // B fragment loader (4 fragments for this wave's 64 columns)
  auto load_b = [&](Frag* b, int k0) {
#pragma unroll
    for (int j = 0; j < 4; ++j) {
      const __bf16* bp = BT + (size_t)(col0 + j * 16 + mr) * K + k0 + half * 16;
      b[j].q[0] = *(const Q16*)(bp);
      b[j].q[1] = *(const Q16*)(bp + 8);
    }
  };

  v8f acc[4] = {};
  Frag bcur[4], bnxt[4];

  const int steps = K / 32;
  stage(0, 0);            // prologue: A k-step 0 -> LDS buffer 0
  load_b(bcur, 0);        // prologue: B k-step 0 -> registers

  for (int s = 0; s < steps; ++s) {
    const int k0  = s * 32;
    const int buf = s & 1;
    stage_wait();          // own staging issue from previous step complete
    __syncthreads();       // everyone's staging for this step visible
    if (s + 1 < steps) {
      stage(k0 + 32, buf ^ 1);   // A: next k-step into other LDS buffer
      load_b(bnxt, k0 + 32);     // B: next k-step into registers
    }

    // A fragment (16x32 bf16) from LDS: lanes 0-15 hold M=mr,
    // K in {0..15}; lanes 16-31 K in {16..31}.
    Frag a;
    const __bf16* ap = &As[(buf * WPM + wm) * TILE_E + mr * LROW + half * 16];
    a.q[0] = *(const Q16*)(ap);
    a.q[1] = *(const Q16*)(ap + 8);

#pragma unroll
    for (int j = 0; j < 4; ++j)
      acc[j] = __builtin_amdgcn_wmma_f32_16x16x32_bf16(
          false, a.v, false, bcur[j].v, (short)0, acc[j], false, false);

#pragma unroll
    for (int j = 0; j < 4; ++j) bcur[j] = bnxt[j];
  }

  // C/D layout: lane<16 -> N=lane, VGPR r -> M=r ; lane>=16 -> M=r+8.
  const int rbase = row0 + half * 8;
#pragma unroll
  for (int j = 0; j < 4; ++j) {
    const int cc = col0 + j * 16 + mr;
    if (cc < Nstore) {
#pragma unroll
      for (int r = 0; r < 8; ++r)
        C[(size_t)(rbase + r) * ldc + cc] = (OutT)acc[j][r];
    }
  }
}

// ---------------------------------------------------------------------------
// Degree / normalization
// ---------------------------------------------------------------------------
__global__ void deg_init_kernel(float* deg, int n) {
  int i = blockIdx.x * blockDim.x + threadIdx.x;
  if (i < n) deg[i] = 1.0f;   // self-loop contribution
}

__global__ void deg_accum_kernel(float* deg, const int* __restrict__ col, int e) {
  int i = blockIdx.x * blockDim.x + threadIdx.x;
  if (i < e) atomicAdd(&deg[col[i]], 1.0f);
}

__global__ void dinv_kernel(float* dinv, const float* __restrict__ deg, int n) {
  int i = blockIdx.x * blockDim.x + threadIdx.x;
  if (i < n) dinv[i] = rsqrtf(deg[i]);
}

// ---------------------------------------------------------------------------
// Casts / padding
// ---------------------------------------------------------------------------
__global__ void cast_x_kernel(const float* __restrict__ x, __bf16* __restrict__ xb,
                              int Mpad, int Mreal, int K) {
  size_t i = (size_t)blockIdx.x * blockDim.x + threadIdx.x;
  if (i >= (size_t)Mpad * K) return;
  int r = (int)(i / (unsigned)K);
  xb[i] = (r < Mreal) ? f2bf(x[i]) : f2bf(0.0f);
}

// WT[Npad x Kpad] = transpose(W[Kin x Nout]) with zero padding
__global__ void cast_wT_kernel(const float* __restrict__ W, __bf16* __restrict__ WT,
                               int Kin, int Kpad, int Nout, int Npad) {
  size_t i = (size_t)blockIdx.x * blockDim.x + threadIdx.x;
  if (i >= (size_t)Npad * Kpad) return;
  int n = (int)(i / (unsigned)Kpad);
  int k = (int)(i % (unsigned)Kpad);
  WT[i] = (n < Nout && k < Kin) ? f2bf(W[(size_t)k * Nout + n]) : f2bf(0.0f);
}

__global__ void relu_cast_kernel(const float* __restrict__ agg, __bf16* __restrict__ hb,
                                 int Mpad, int Mreal, int F, int ld) {
  size_t i = (size_t)blockIdx.x * blockDim.x + threadIdx.x;
  if (i >= (size_t)Mpad * ld) return;
  int r = (int)(i / (unsigned)ld);
  int j = (int)(i % (unsigned)ld);
  float v = (r < Mreal && j < F) ? agg[(size_t)r * F + j] : 0.0f;
  hb[i] = f2bf(fmaxf(v, 0.0f));
}

__global__ void init_bias_kernel(float* __restrict__ o, const float* __restrict__ b,
                                 int M, int F) {
  size_t i = (size_t)blockIdx.x * blockDim.x + threadIdx.x;
  if (i >= (size_t)M * F) return;
  o[i] = b[i % (unsigned)F];
}

// ---------------------------------------------------------------------------
// Gather-normalize-scatter aggregation: one wave per (edge or self-loop).
// agg[col] += h[row] * dinv[row]*dinv[col].  4 features per lane per iter.
// F must be a multiple of 4.
// ---------------------------------------------------------------------------
struct U2pod { unsigned int a, b; };          // 4 x bf16
struct alignas(16) F4pod { float x, y, z, w; };

__global__ void agg_scatter_bf16_kernel(const __bf16* __restrict__ h,
                                        const int* __restrict__ row,
                                        const int* __restrict__ col,
                                        const float* __restrict__ dinv,
                                        float* __restrict__ agg,
                                        int E, int N, int F, int ldh, int lda) {
  int wid  = (int)(((size_t)blockIdx.x * blockDim.x + threadIdx.x) >> 5);
  int lane = threadIdx.x & 31;
  if (wid >= E + N) return;
  int r, c;
  if (wid < E) { r = row[wid]; c = col[wid]; }
  else         { r = c = wid - E; }              // self-loop
  float nrm = dinv[r] * dinv[c];
  const __bf16* hr = h + (size_t)r * ldh;
  float* ar        = agg + (size_t)c * lda;
  for (int f = lane * 4; f < F; f += 128) {
    union { U2pod u; __bf16 e[4]; } pk;
    pk.u = *(const U2pod*)(hr + f);
#pragma unroll
    for (int k = 0; k < 4; ++k)
      atomicAdd(&ar[f + k], (float)pk.e[k] * nrm);
  }
}

__global__ void agg_scatter_f32_kernel(const float* __restrict__ h,
                                       const int* __restrict__ row,
                                       const int* __restrict__ col,
                                       const float* __restrict__ dinv,
                                       float* __restrict__ agg,
                                       int E, int N, int F, int ldh, int lda) {
  int wid  = (int)(((size_t)blockIdx.x * blockDim.x + threadIdx.x) >> 5);
  int lane = threadIdx.x & 31;
  if (wid >= E + N) return;
  int r, c;
  if (wid < E) { r = row[wid]; c = col[wid]; }
  else         { r = c = wid - E; }              // self-loop
  float nrm = dinv[r] * dinv[c];
  const float* hr = h + (size_t)r * ldh;
  float* ar       = agg + (size_t)c * lda;
  for (int f = lane * 4; f < F; f += 128) {
    F4pod v = *(const F4pod*)(hr + f);
    atomicAdd(&ar[f + 0], v.x * nrm);
    atomicAdd(&ar[f + 1], v.y * nrm);
    atomicAdd(&ar[f + 2], v.z * nrm);
    atomicAdd(&ar[f + 3], v.w * nrm);
  }
}

// ---------------------------------------------------------------------------
extern "C" void kernel_launch(void* const* d_in, const int* in_sizes, int n_in,
                              void* d_out, int out_size, void* d_ws, size_t ws_size,
                              hipStream_t stream) {
  const float* x  = (const float*)d_in[0];
  const int*   ei = (const int*)d_in[1];
  const float* W1 = (const float*)d_in[2];
  const float* b1 = (const float*)d_in[3];
  const float* W2 = (const float*)d_in[4];
  const float* b2 = (const float*)d_in[5];
  float* out = (float*)d_out;

  const int* row = ei;        // edge_index[0]
  const int* col = ei + EE;   // edge_index[1]

  // Workspace carve-up (256B aligned)
  char* p = (char*)d_ws;
  auto alloc = [&](size_t bytes) -> char* {
    char* r = p;
    p += (bytes + 255) & ~(size_t)255;
    return r;
  };
  float*  deg  = (float*) alloc((size_t)NN * 4);
  float*  dinv = (float*) alloc((size_t)NN * 4);
  __bf16* xb   = (__bf16*)alloc((size_t)MPAD * INF_ * 2);
  __bf16* w1t  = (__bf16*)alloc((size_t)HIDPAD * INF_ * 2);
  __bf16* h1   = (__bf16*)alloc((size_t)MPAD * HID * 2);
  float*  agg1 = (float*) alloc((size_t)NN * HID * 4);
  __bf16* hr   = (__bf16*)alloc((size_t)MPAD * KPAD2 * 2);
  __bf16* w2t  = (__bf16*)alloc((size_t)OUTF * KPAD2 * 2);
  float*  h2   = (float*) alloc((size_t)MPAD * OUTF * 4);

  const int T = 256;
  auto blocks = [](size_t n, int t) { return (unsigned)((n + t - 1) / t); };

  // --- normalization coefficients ---
  deg_init_kernel  <<<blocks(NN, T), T, 0, stream>>>(deg, NN);
  deg_accum_kernel <<<blocks(EE, T), T, 0, stream>>>(deg, col, EE);
  dinv_kernel      <<<blocks(NN, T), T, 0, stream>>>(dinv, deg, NN);

  // --- layer 1: cast, GEMM (WMMA + async LDS), bias-init, aggregate ---
  cast_x_kernel <<<blocks((size_t)MPAD * INF_, T), T, 0, stream>>>(x, xb, MPAD, NN, INF_);
  cast_wT_kernel<<<blocks((size_t)HIDPAD * INF_, T), T, 0, stream>>>(W1, w1t, INF_, INF_, HID, HIDPAD);

  // GEMM1: WPM=1, WPN=8, async-LDS A staging. Block tile 16x512.
  {
    unsigned g = (unsigned)((MPAD / 16) * (HIDPAD / 512));
    gemm_bf16_async<1, 8, false, __bf16><<<g, 256, 0, stream>>>(
        xb, w1t, h1, MPAD, HIDPAD, HID, INF_, HID);
  }

  init_bias_kernel<<<blocks((size_t)NN * HID, T), T, 0, stream>>>(agg1, b1, NN, HID);

  int nedges = EE + NN;
  agg_scatter_bf16_kernel<<<(nedges + 7) / 8, 256, 0, stream>>>(
      h1, row, col, dinv, agg1, EE, NN, HID, HID, HID);

  // --- layer 2: relu+cast, GEMM (WMMA + TDM A staging), bias-init, agg ---
  relu_cast_kernel<<<blocks((size_t)MPAD * KPAD2, T), T, 0, stream>>>(agg1, hr, MPAD, NN, HID, KPAD2);
  cast_wT_kernel  <<<blocks((size_t)OUTF * KPAD2, T), T, 0, stream>>>(W2, w2t, HID, KPAD2, OUTF, OUTF);

  // GEMM2: WPM=4, WPN=2, TDM A staging. Block tile 64x128.
  {
    unsigned g = (unsigned)((MPAD / 64) * (OUTF / 128));
    gemm_bf16_async<4, 2, true, float><<<g, 256, 0, stream>>>(
        hr, w2t, h2, MPAD, OUTF, OUTF, KPAD2, OUTF);
  }

  init_bias_kernel<<<blocks((size_t)NN * OUTF, T), T, 0, stream>>>(out, b2, NN, OUTF);

  agg_scatter_f32_kernel<<<(nedges + 7) / 8, 256, 0, stream>>>(
      h2, row, col, dinv, out, EE, NN, OUTF, OUTF, OUTF);
}
